// IterativePathFinder_71777493451385
// MI455X (gfx1250) — matrix operand order repaired and checked by reading, not compiled, not run on hardware
//
#include <hip/hip_runtime.h>
#include <math.h>

typedef __attribute__((ext_vector_type(16))) __bf16 v16bf;
typedef __attribute__((ext_vector_type(8)))  float  v8f;
typedef unsigned int u32;

#define DD   768
#define HID  192
#define BB   8
#define LQN  64
#define NCN  2048
#define NEN  1200
#define KCE  1536   // 2*D
#define OUTW 3248   // NC + NE

static __device__ __forceinline__ __bf16 tobf(float f) {
  u32 u = __builtin_bit_cast(u32, f);
  u32 r = (u + 0x7FFFu + ((u >> 16) & 1u)) >> 16;
  return __builtin_bit_cast(__bf16, (unsigned short)r);
}

static __device__ __forceinline__ float blockReduceSum(float v, float* red) {
  int tid = threadIdx.x;
  red[tid] = v; __syncthreads();
  for (int s = blockDim.x >> 1; s > 0; s >>= 1) {
    if (tid < s) red[tid] += red[tid + s];
    __syncthreads();
  }
  return red[0];
}

// ---------------- packing kernels ----------------
// Pack row-major [K][N] fp32 weights into WMMA B-fragment order:
// dst[((kb*NT + nt)*32 + lane)*16 + e] = bf16(src[(kb*32 + e + (lane>=16?16:0))*N + nt*16 + (lane&15)])
__global__ __launch_bounds__(256) void ipf_pack(const float* __restrict__ src,
                                                __bf16* __restrict__ dst,
                                                int N, int total) {
  int g = blockIdx.x * 256 + threadIdx.x;
  if (g >= total) return;
  int e    = g & 15;
  int lane = (g >> 4) & 31;
  int NT   = N >> 4;
  int nt   = (g >> 9) % NT;
  int kb   = g / (512 * NT);
  int k = kb * 32 + e + ((lane & 16) ? 16 : 0);
  int n = nt * 16 + (lane & 15);
  dst[g] = tobf(src[(size_t)k * N + n]);
}

// Pack q_hidden[b][l][k] as B-fragments of q^T (K=768, N=64) per batch:
// dst[(((b*24 + kb)*4 + nt)*32 + lane)*16 + e]
__global__ __launch_bounds__(256) void ipf_pack_q(const float* __restrict__ q,
                                                  __bf16* __restrict__ dst) {
  int g = blockIdx.x * 256 + threadIdx.x;
  if (g >= BB * DD * LQN) return;
  int e    = g & 15;
  int lane = (g >> 4) & 31;
  int nt   = (g >> 9) & 3;        // N/16 = 4
  int kb   = (g >> 11) % 24;      // K/32 = 24
  int b    = g / (DD * LQN);
  int k = kb * 32 + e + ((lane & 16) ? 16 : 0);
  int l = nt * 16 + (lane & 15);
  dst[g] = tobf(q[((size_t)b * LQN + l) * DD + k]);
}

// q_emb = max_l(q*mask)
__global__ __launch_bounds__(256) void ipf_prepA(const float* __restrict__ q,
                                                 const int* __restrict__ mask,
                                                 float* __restrict__ q_emb) {
  int g = blockIdx.x * 256 + threadIdx.x;        // g in [0, B*D)
  if (g >= BB * DD) return;
  int b = g / DD, d = g % DD;
  float m = -INFINITY;
  for (int l = 0; l < LQN; ++l) {
    float qv = q[(size_t)(b * LQN + l) * DD + d];
    float mk = mask[b * LQN + l] ? 1.0f : 0.0f;
    m = fmaxf(m, qv * mk);
  }
  q_emb[g] = m;
}

// mem2, c_query, t_query, memory_dot (small GEMVs)
__global__ __launch_bounds__(256) void ipf_small(const float* __restrict__ q,
                                                 const float* __restrict__ q_emb,
                                                 const float* __restrict__ W_mem2,
                                                 const float* __restrict__ b_mem2,
                                                 const float* __restrict__ W_cs,
                                                 const float* __restrict__ b_cs,
                                                 const float* __restrict__ W_ts,
                                                 const float* __restrict__ b_ts,
                                                 const float* __restrict__ w_mem1,
                                                 float* __restrict__ mem2,
                                                 float* __restrict__ c_query,
                                                 float* __restrict__ t_query,
                                                 float* __restrict__ memory_dot) {
  int g = blockIdx.x * 256 + threadIdx.x;
  const int NM2 = BB * LQN * HID;     // 98304
  const int NCQ = BB * 2 * DD;        // 12288
  const int NTQ = BB * 5 * DD;        // 30720
  if (g < NM2) {
    int h = g % HID, bl = g / HID;
    float s = b_mem2[h];
    for (int k = 0; k < DD; ++k) s += q[(size_t)bl * DD + k] * W_mem2[(size_t)k * HID + h];
    mem2[g] = s;
  } else if (g < NM2 + NCQ) {
    int gg = g - NM2; int j = gg % (2 * DD), b = gg / (2 * DD);
    float s = b_cs[j];
    for (int k = 0; k < DD; ++k) s += q_emb[b * DD + k] * W_cs[(size_t)k * (2 * DD) + j];
    c_query[gg] = s;
  } else if (g < NM2 + NCQ + NTQ) {
    int gg = g - NM2 - NCQ; int j = gg % (5 * DD), b = gg / (5 * DD);
    float s = b_ts[j];
    for (int k = 0; k < DD; ++k) s += q_emb[b * DD + k] * W_ts[(size_t)k * (5 * DD) + j];
    t_query[gg] = s;
  } else if (g < NM2 + NCQ + NTQ + BB * LQN) {
    int gg = g - NM2 - NCQ - NTQ;     // b*LQ+l
    float s = 0.f;
    for (int k = 0; k < DD; ++k) s += q[(size_t)gg * DD + k] * w_mem1[k];
    memory_dot[gg] = s;
  }
}

// Store (r,k) of a 16x32 A tile into fragment order: lane = r + (k>=16 ? 16:0),
// elem = k&15. Each lane's 16 bf16 are then contiguous (32B).
#define ATP_STORE(AtP, r, k, val) \
  AtP[(r) + (((k) >> 4) & 1) * 16][(k) & 15] = (val)

// ---------------- GEMM 1: cf = gather(concept,dist) @ W_ce + b_ce (+input_dot) ----------------
__global__ __launch_bounds__(256) void ipf_gemm_cf(const int* __restrict__ head_cids,
                                                   const int* __restrict__ distances,
                                                   const float* __restrict__ concept_table,
                                                   const float* __restrict__ dist_table,
                                                   const __bf16* __restrict__ Wce_pk,
                                                   const float* __restrict__ b_ce,
                                                   const float* __restrict__ w_in1,
                                                   float* __restrict__ cf,
                                                   float* __restrict__ input_dot) {
  __shared__ alignas(32) __bf16 AtP[32][16];
  __shared__ int cid_s[16], dst_s[16];
  __shared__ float idotS[16];
  int tid = threadIdx.x;
  int m0 = blockIdx.x * 16;                       // global row (= b*NC + n)
  if (tid < 16) {
    cid_s[tid] = head_cids[m0 + tid];
    dst_s[tid] = distances[m0 + tid];
    idotS[tid] = 0.f;
  }
  __syncthreads();
  int wid = tid >> 5, lane = tid & 31, l16 = lane & 15;
  bool top = lane < 16;
  const v16bf* Wp = (const v16bf*)Wce_pk;
  v8f acc[6];
  #pragma unroll
  for (int t = 0; t < 6; ++t)
    #pragma unroll
    for (int v = 0; v < 8; ++v) acc[t][v] = 0.f;

  for (int kb = 0; kb < KCE / 32; ++kb) {
    for (int i = tid; i < 512; i += 256) {        // A tile 16x32, gather + cvt
      int r = i >> 5, k = i & 31, kg = kb * 32 + k;
      float val = (kg < DD) ? concept_table[(size_t)cid_s[r] * DD + kg]
                            : dist_table[(size_t)dst_s[r] * DD + (kg - DD)];
      ATP_STORE(AtP, r, k, tobf(val));
    }
    __syncthreads();
    v16bf a = *(const v16bf*)(&AtP[lane][0]);
    int kbn = (kb + 1 < KCE / 32) ? kb + 1 : kb;
    __builtin_prefetch((const void*)&Wp[((size_t)kbn * 48 + wid * 6) * 32 + lane], 0, 3);
    #pragma unroll
    for (int t = 0; t < 6; ++t) {
      v16bf bq = Wp[((size_t)kb * 48 + wid * 6 + t) * 32 + lane];
      acc[t] = __builtin_amdgcn_wmma_f32_16x16x32_bf16(false, a, false, bq, (short)0,
                                                       acc[t], false, false);
    }
    __syncthreads();
  }
  float bias[6], w1v[6];
  #pragma unroll
  for (int t = 0; t < 6; ++t) {
    int n = wid * 96 + t * 16 + l16;
    bias[t] = b_ce[n];
    w1v[t] = w_in1[n];
  }
  #pragma unroll
  for (int v = 0; v < 8; ++v) {
    int row = m0 + v + (top ? 0 : 8);
    float p = 0.f;
    #pragma unroll
    for (int t = 0; t < 6; ++t) {
      int n = wid * 96 + t * 16 + l16;
      float cv = acc[t][v] + bias[t];
      cf[(size_t)row * DD + n] = cv;
      p += cv * w1v[t];
    }
    atomicAdd(&idotS[v + (top ? 0 : 8)], p);
  }
  __syncthreads();
  if (tid < 16) input_dot[m0 + tid] = idotS[tid];
}

// ---------------- GEMM 2: inp2 = cf @ W_in2 + b_in2 ----------------
__global__ __launch_bounds__(128) void ipf_gemm_inp2(const float* __restrict__ cf,
                                                     const __bf16* __restrict__ Win2_pk,
                                                     const float* __restrict__ b_in2,
                                                     float* __restrict__ inp2) {
  __shared__ alignas(32) __bf16 AtP[32][16];
  int tid = threadIdx.x;
  int m0 = blockIdx.x * 16;
  int wid = tid >> 5, lane = tid & 31, l16 = lane & 15;
  bool top = lane < 16;
  const v16bf* Wp = (const v16bf*)Win2_pk;
  v8f acc[3];
  #pragma unroll
  for (int t = 0; t < 3; ++t)
    #pragma unroll
    for (int v = 0; v < 8; ++v) acc[t][v] = 0.f;

  for (int kb = 0; kb < DD / 32; ++kb) {
    for (int i = tid; i < 512; i += 128) {
      int r = i >> 5, k = i & 31;
      ATP_STORE(AtP, r, k, tobf(cf[(size_t)(m0 + r) * DD + kb * 32 + k]));
    }
    __syncthreads();
    v16bf a = *(const v16bf*)(&AtP[lane][0]);
    int kbn = (kb + 1 < DD / 32) ? kb + 1 : kb;
    __builtin_prefetch((const void*)&Wp[((size_t)kbn * 12 + wid * 3) * 32 + lane], 0, 3);
    #pragma unroll
    for (int t = 0; t < 3; ++t) {
      v16bf bq = Wp[((size_t)kb * 12 + wid * 3 + t) * 32 + lane];
      acc[t] = __builtin_amdgcn_wmma_f32_16x16x32_bf16(false, a, false, bq, (short)0,
                                                       acc[t], false, false);
    }
    __syncthreads();
  }
  #pragma unroll
  for (int t = 0; t < 3; ++t) {
    int n = wid * 48 + t * 16 + l16;
    float b = b_in2[n];
    #pragma unroll
    for (int v = 0; v < 8; ++v) {
      int row = m0 + v + (top ? 0 : 8);
      inp2[(size_t)row * HID + n] = acc[t][v] + b;
    }
  }
}

// ---------------- GEMM 3: cross attention + softmax + out1 + rowmax ----------------
__global__ __launch_bounds__(128) void ipf_gemm_att(const float* __restrict__ cf,
                                                    const __bf16* __restrict__ Qpk,
                                                    const float* __restrict__ input_dot,
                                                    const float* __restrict__ memory_dot,
                                                    const int* __restrict__ mask,
                                                    const float* __restrict__ mem2,
                                                    float* __restrict__ out1,
                                                    float* __restrict__ rowmax) {
  __shared__ alignas(32) __bf16 AtP[32][16];
  __shared__ float attS[16][LQN];
  __shared__ float mem2S[LQN * HID];
  __shared__ float idotS[16];
  __shared__ float mdl[LQN];
  const float INVSQ = 0.036084391824351614f;      // 1/sqrt(768)
  int tid = threadIdx.x;
  int b = blockIdx.y, n0 = blockIdx.x * 16;
  int wid = tid >> 5, lane = tid & 31, l16 = lane & 15;
  bool top = lane < 16;

  for (int i = tid; i < LQN * HID; i += 128) mem2S[i] = mem2[(size_t)b * LQN * HID + i];
  if (tid < 16) idotS[tid] = input_dot[b * NCN + n0 + tid];
  if (tid < LQN) {
    float mk = mask[b * LQN + tid] ? 1.0f : 0.0f;
    mdl[tid] = memory_dot[b * LQN + tid] - 1e30f * (1.0f - mk);
  }
  const v16bf* Qp = (const v16bf*)Qpk;
  v8f acc;
  #pragma unroll
  for (int v = 0; v < 8; ++v) acc[v] = 0.f;

  for (int kb = 0; kb < DD / 32; ++kb) {
    for (int i = tid; i < 512; i += 128) {
      int r = i >> 5, k = i & 31;
      ATP_STORE(AtP, r, k, tobf(cf[((size_t)b * NCN + n0 + r) * DD + kb * 32 + k]));
    }
    __syncthreads();
    v16bf a = *(const v16bf*)(&AtP[lane][0]);
    v16bf bq = Qp[(((size_t)b * 24 + kb) * 4 + wid) * 32 + lane];
    acc = __builtin_amdgcn_wmma_f32_16x16x32_bf16(false, a, false, bq, (short)0,
                                                  acc, false, false);
    __syncthreads();
  }
  int col = wid * 16 + l16;
  #pragma unroll
  for (int v = 0; v < 8; ++v) {
    int r = v + (top ? 0 : 8);
    attS[r][col] = acc[v] * INVSQ + idotS[r] + mdl[col];
  }
  __syncthreads();
  if (tid < 16) {
    float mx = -INFINITY;
    for (int l = 0; l < LQN; ++l) mx = fmaxf(mx, attS[tid][l]);
    rowmax[b * NCN + n0 + tid] = mx;
    float sum = 0.f;
    for (int l = 0; l < LQN; ++l) sum += expf(attS[tid][l] - mx);
    float inv = 1.0f / sum;
    for (int l = 0; l < LQN; ++l) attS[tid][l] = expf(attS[tid][l] - mx) * inv;
  }
  __syncthreads();
  for (int o = tid; o < 16 * HID; o += 128) {
    int r = o / HID, h = o % HID;
    float s = 0.f;
    for (int l = 0; l < LQN; ++l) s += attS[r][l] * mem2S[l * HID + h];
    out1[((size_t)b * NCN + n0 + r) * HID + h] = s;
  }
}

// ---------------- out2: w2 = softmax_n(rowmax), out2 = sum_n w2*inp2 ----------------
__global__ __launch_bounds__(256) void ipf_out2(const float* __restrict__ rowmax,
                                                const float* __restrict__ inp2,
                                                float* __restrict__ out2) {
  __shared__ float red[256];
  __shared__ float s_max, s_sum;
  int b = blockIdx.x, tid = threadIdx.x;
  float m = -INFINITY;
  for (int n = tid; n < NCN; n += 256) m = fmaxf(m, rowmax[b * NCN + n]);
  red[tid] = m; __syncthreads();
  for (int s = 128; s > 0; s >>= 1) {
    if (tid < s) red[tid] = fmaxf(red[tid], red[tid + s]);
    __syncthreads();
  }
  if (tid == 0) s_max = red[0];
  __syncthreads();
  float sm = 0.f;
  for (int n = tid; n < NCN; n += 256) sm += expf(rowmax[b * NCN + n] - s_max);
  float tot = blockReduceSum(sm, red);
  if (tid == 0) s_sum = tot;
  __syncthreads();
  if (tid < HID) {
    float a = 0.f;
    for (int n = 0; n < NCN; ++n)
      a += expf(rowmax[b * NCN + n] - s_max) * inp2[((size_t)b * NCN + n) * HID + tid];
    out2[b * HID + tid] = a / s_sum;
  }
}

// ---------------- concept logits ----------------
__global__ __launch_bounds__(256) void ipf_clogits(const float* __restrict__ cf,
                                                   const float* __restrict__ inp2,
                                                   const float* __restrict__ out1,
                                                   const float* __restrict__ out2,
                                                   const float* __restrict__ c_query,
                                                   float* __restrict__ out) {
  __shared__ float red[256];
  int row = blockIdx.x, tid = threadIdx.x;
  int b = row / NCN, n = row % NCN;
  const float* cq = c_query + (size_t)b * (2 * DD);
  float p = 0.f;
  for (int k = tid; k < DD; k += 256) p += cf[(size_t)row * DD + k] * cq[k];
  if (tid < HID) {
    float i2 = inp2[(size_t)row * HID + tid];
    float o1 = out1[(size_t)row * HID + tid];
    float o2 = out2[b * HID + tid];
    p += i2 * cq[DD + tid] + o1 * cq[DD + HID + tid]
       + i2 * o1 * cq[DD + 2 * HID + tid] + o2 * o1 * cq[DD + 3 * HID + tid];
  }
  float s = blockReduceSum(p, red);
  if (tid == 0) out[(size_t)b * OUTW + n] = 1.0f / (1.0f + expf(-s));
}

// ---------------- triple logits ----------------
__global__ __launch_bounds__(256) void ipf_tlogits(const float* __restrict__ cf,
                                                   const float* __restrict__ inp2,
                                                   const float* __restrict__ out1,
                                                   const float* __restrict__ out2,
                                                   const float* __restrict__ rel_table,
                                                   const float* __restrict__ t_query,
                                                   const int* __restrict__ head_idxs,
                                                   const int* __restrict__ tail_idxs,
                                                   const int* __restrict__ rel_ids,
                                                   float* __restrict__ out) {
  __shared__ float red[256];
  int idx = blockIdx.x, tid = threadIdx.x;
  int b = idx / NEN, e = idx % NEN;
  int hix = head_idxs[b * NEN + e];
  int tix = tail_idxs[b * NEN + e];
  int rix = rel_ids[b * NEN + e];
  const float* hrow = cf + ((size_t)b * NCN + hix) * DD;
  const float* trow = cf + ((size_t)b * NCN + tix) * DD;
  const float* rrow = rel_table + (size_t)rix * DD;
  const float* tq = t_query + (size_t)b * (5 * DD);
  float p = 0.f;
  for (int k = tid; k < DD; k += 256) {
    float hv = hrow[k], tv = trow[k];
    p += hv * tq[k] + rrow[k] * tq[DD + k] + tv * tq[2 * DD + k] + hv * tv * tq[3 * DD + k];
  }
  if (tid < HID) {
    float i2 = inp2[((size_t)b * NCN + hix) * HID + tid];
    float o1 = out1[((size_t)b * NCN + hix) * HID + tid];
    float o2 = out2[b * HID + tid];
    p += i2 * tq[4 * DD + tid] + o1 * tq[4 * DD + HID + tid]
       + i2 * o1 * tq[4 * DD + 2 * HID + tid] + o2 * o1 * tq[4 * DD + 3 * HID + tid];
  }
  float s = blockReduceSum(p, red);
  if (tid == 0) out[(size_t)b * OUTW + NCN + e] = 1.0f / (1.0f + expf(-s));
}

// ---------------- host launcher ----------------
extern "C" void kernel_launch(void* const* d_in, const int* in_sizes, int n_in,
                              void* d_out, int out_size, void* d_ws, size_t ws_size,
                              hipStream_t stream) {
  const float* q             = (const float*)d_in[0];
  const int*   mask          = (const int*)d_in[1];
  const int*   head_cids     = (const int*)d_in[2];
  const int*   distances     = (const int*)d_in[3];
  const int*   head_idxs     = (const int*)d_in[4];
  const int*   tail_idxs     = (const int*)d_in[5];
  const int*   rel_ids       = (const int*)d_in[6];
  const float* concept_table = (const float*)d_in[7];
  const float* dist_table    = (const float*)d_in[8];
  const float* rel_table     = (const float*)d_in[9];
  const float* W_ce          = (const float*)d_in[10];
  const float* b_ce          = (const float*)d_in[11];
  const float* W_cs          = (const float*)d_in[12];
  const float* b_cs          = (const float*)d_in[13];
  const float* W_ts          = (const float*)d_in[14];
  const float* b_ts          = (const float*)d_in[15];
  const float* w_in1         = (const float*)d_in[16];
  const float* w_mem1        = (const float*)d_in[17];
  const float* W_in2         = (const float*)d_in[18];
  const float* b_in2         = (const float*)d_in[19];
  const float* W_mem2        = (const float*)d_in[20];
  const float* b_mem2        = (const float*)d_in[21];
  float* out = (float*)d_out;

  char* ws = (char*)d_ws;
  size_t off = 0;
  auto alloc = [&](size_t bytes) {
    char* p = ws + off;
    off = (off + bytes + 255) & ~(size_t)255;
    return p;
  };
  __bf16* Wce_pk   = (__bf16*)alloc((size_t)KCE * DD * 2);
  __bf16* Win2_pk  = (__bf16*)alloc((size_t)DD * HID * 2);
  __bf16* Qpk      = (__bf16*)alloc((size_t)BB * DD * LQN * 2);
  float*  cf       = (float*)alloc((size_t)BB * NCN * DD * 4);
  float*  inp2     = (float*)alloc((size_t)BB * NCN * HID * 4);
  float*  out1     = (float*)alloc((size_t)BB * NCN * HID * 4);
  float*  q_emb    = (float*)alloc((size_t)BB * DD * 4);
  float*  mdot     = (float*)alloc((size_t)BB * LQN * 4);
  float*  mem2     = (float*)alloc((size_t)BB * LQN * HID * 4);
  float*  c_query  = (float*)alloc((size_t)BB * 2 * DD * 4);
  float*  t_query  = (float*)alloc((size_t)BB * 5 * DD * 4);
  float*  indot    = (float*)alloc((size_t)BB * NCN * 4);
  float*  rowmax   = (float*)alloc((size_t)BB * NCN * 4);
  float*  out2v    = (float*)alloc((size_t)BB * HID * 4);

  ipf_pack<<<(KCE * DD + 255) / 256, 256, 0, stream>>>(W_ce, Wce_pk, DD, KCE * DD);
  ipf_pack<<<(DD * HID + 255) / 256, 256, 0, stream>>>(W_in2, Win2_pk, HID, DD * HID);
  ipf_pack_q<<<(BB * DD * LQN + 255) / 256, 256, 0, stream>>>(q, Qpk);
  ipf_prepA<<<(BB * DD + 255) / 256, 256, 0, stream>>>(q, mask, q_emb);
  {
    int tot = BB * LQN * HID + BB * 2 * DD + BB * 5 * DD + BB * LQN;
    ipf_small<<<(tot + 255) / 256, 256, 0, stream>>>(q, q_emb, W_mem2, b_mem2, W_cs, b_cs,
                                                     W_ts, b_ts, w_mem1, mem2, c_query,
                                                     t_query, mdot);
  }
  ipf_gemm_cf<<<(BB * NCN) / 16, 256, 0, stream>>>(head_cids, distances, concept_table,
                                                   dist_table, Wce_pk, b_ce, w_in1,
                                                   cf, indot);
  ipf_gemm_inp2<<<(BB * NCN) / 16, 128, 0, stream>>>(cf, Win2_pk, b_in2, inp2);
  ipf_gemm_att<<<dim3(NCN / 16, BB), 128, 0, stream>>>(cf, Qpk, indot, mdot, mask, mem2,
                                                       out1, rowmax);
  ipf_out2<<<BB, 256, 0, stream>>>(rowmax, inp2, out2v);
  ipf_clogits<<<BB * NCN, 256, 0, stream>>>(cf, inp2, out1, out2v, c_query, out);
  ipf_tlogits<<<BB * NEN, 256, 0, stream>>>(cf, inp2, out1, out2v, rel_table, t_query,
                                            head_idxs, tail_idxs, rel_ids, out);
}
